// DcnPool_66812511257279
// MI455X (gfx1250) — matrix operand order repaired and checked by reading, not compiled, not run on hardware
//
#include <hip/hip_runtime.h>
#include <hip/hip_bf16.h>
#include <math.h>

typedef __bf16 bf16;
typedef __attribute__((ext_vector_type(16))) __bf16 v16bf;
typedef __attribute__((ext_vector_type(8)))  float   v8f;

namespace dcnpool {

constexpr int kB  = 2;
constexpr int kC  = 256;
constexpr int kH  = 128;
constexpr int kW  = 128;
constexpr int kHW = kH * kW;                 // 16384 = 2^14
constexpr int kP  = kB * kHW;                // 32768 = 2^15
constexpr size_t kConvOut = (size_t)kB * kC * kHW;  // 8388608
constexpr float kEps = 1e-5f;

constexpr int kPitch = 40;                   // LDS tile pitch (bf16 elems), 80 B rows

union FragU {
  uint4 q[2];
  v16bf v;
};

__device__ inline v8f vzero8() {
  v8f z;
#pragma unroll
  for (int i = 0; i < 8; ++i) z[i] = 0.f;
  return z;
}

// Issue a 32-byte global -> LDS async copy (2 x b128) for one lane.
//   ldsOff : byte offset of destination inside the block's LDS allocation
//   src    : global source (16-byte aligned, 32 bytes copied)
__device__ inline void async_copy32(unsigned ldsOff, const bf16* src) {
  unsigned long long ga = (unsigned long long)(uintptr_t)src;
  asm volatile("global_load_async_to_lds_b128 %0, %1, off"
               :: "v"(ldsOff), "v"(ga) : "memory");
  asm volatile("global_load_async_to_lds_b128 %0, %1, off offset:16"
               :: "v"(ldsOff), "v"(ga) : "memory");
}

__device__ inline void async_wait0() {
  asm volatile("s_wait_asynccnt 0x0" ::: "memory");
}

// ---------------------------------------------------------------------------
// Implicit-GEMM 3x3 conv:  out[m][p] = sum_{t,c} wpk[t][m][c] * x[c][shift_t(p)]
//   wpk : [9][Mpad][Cin] bf16 (tap, cout, cin)   (Mpad = 64 or 256)
//   xsrc: [Cin][P] bf16, p = b*16384 + y*128 + x, zero pad outside image
//   out : [Mpad][P] f32
// Block: 128 threads (4 waves). Tile: 64 (M) x 64 (N). Wave quadrant 32x32.
// LDS tiles are K-innermost so WMMA fragments load as 2 x ds_load_b128.
// ---------------------------------------------------------------------------
__global__ __launch_bounds__(128)
void conv3x3_wmma_kernel(const bf16* __restrict__ wpk,
                         const bf16* __restrict__ xsrc,
                         float* __restrict__ out,
                         int Cin, int Mpad)
{
  __shared__ bf16 smem[2 * 64 * kPitch];
  bf16* aTp = smem;                 // [64 m][kPitch] (k inner)
  bf16* bTp = smem + 64 * kPitch;   // [64 n][kPitch] (k inner)

  const int tid   = threadIdx.x;
  const int wv    = tid >> 5;
  const int lane  = tid & 31;
  const int half  = lane >> 4;
  const int lan15 = lane & 15;

  const int n0 = blockIdx.x * 64;   // pixel tile base
  const int m0 = blockIdx.y * 64;   // out-channel tile base

  const int bImg = n0 >> 14;
  const int yx   = n0 & (kHW - 1);
  const int yRow = yx >> 7;
  const int x0   = yx & (kW - 1);

  const int wM = (wv >> 1) * 32;
  const int wN = (wv & 1) * 32;

  v8f acc[2][2];
#pragma unroll
  for (int i = 0; i < 2; ++i)
#pragma unroll
    for (int j = 0; j < 2; ++j) acc[i][j] = vzero8();

  // A staging: 64 rows x 32 k, thread -> (row, 16-elem k segment)
  const int mRow = tid >> 1;
  const int kSeg = (tid & 1) * 16;
  // B staging: 32 k rows x 64 n, thread -> (k row, 16-pixel segment)
  const int kB_  = tid >> 2;
  const int nA   = (tid & 3) * 16;

  for (int t = 0; t < 9; ++t) {
    const int ki = t / 3;
    const int kj = t % 3;
    const int py = yRow + ki - 1;
    const bool rowOK = (py >= 0) && (py < kH);
    for (int kc = 0; kc < Cin; kc += 32) {
      // ---- stage A (weights) via async global->LDS copy ----
      {
        const bf16* s = wpk + ((size_t)(t * Mpad + m0 + mRow) * Cin + kc + kSeg);
        async_copy32((unsigned)((mRow * kPitch + kSeg) * 2), s);
      }
      // ---- stage B transposed ([n][k]) with spatial shift + zero pad ----
      {
        const bf16 z = (bf16)0.0f;
        if (rowOK) {
          const bf16* s = xsrc + ((size_t)(kc + kB_) * kP + (bImg << 14) + (py << 7));
#pragma unroll
          for (int e = 0; e < 16; ++e) {
            int px = x0 + nA + e + kj - 1;
            bTp[(nA + e) * kPitch + kB_] = (px >= 0 && px < kW) ? s[px] : z;
          }
        } else {
#pragma unroll
          for (int e = 0; e < 16; ++e) bTp[(nA + e) * kPitch + kB_] = z;
        }
      }
      async_wait0();
      __syncthreads();

      // ---- fragments: contiguous K-runs -> 2 x b128 per fragment ----
      v16bf af[2], bfg[2];
#pragma unroll
      for (int mi = 0; mi < 2; ++mi) {
        const int mrow = wM + mi * 16 + lan15;
        FragU u;
        u.q[0] = *reinterpret_cast<const uint4*>(&aTp[mrow * kPitch + half * 8]);
        u.q[1] = *reinterpret_cast<const uint4*>(&aTp[mrow * kPitch + 16 + half * 8]);
        af[mi] = u.v;
      }
#pragma unroll
      for (int ni = 0; ni < 2; ++ni) {
        const int ncol = wN + ni * 16 + lan15;
        FragU u;
        u.q[0] = *reinterpret_cast<const uint4*>(&bTp[ncol * kPitch + half * 8]);
        u.q[1] = *reinterpret_cast<const uint4*>(&bTp[ncol * kPitch + 16 + half * 8]);
        bfg[ni] = u.v;
      }
#pragma unroll
      for (int mi = 0; mi < 2; ++mi)
#pragma unroll
        for (int ni = 0; ni < 2; ++ni)
          acc[mi][ni] = __builtin_amdgcn_wmma_f32_16x16x32_bf16(
              false, af[mi], false, bfg[ni], (short)0, acc[mi][ni], false, false);
      __syncthreads();
    }
  }

#pragma unroll
  for (int mi = 0; mi < 2; ++mi) {
#pragma unroll
    for (int ni = 0; ni < 2; ++ni) {
      const int gn = n0 + wN + ni * 16 + lan15;
      const int gm = m0 + wM + mi * 16 + half * 8;
#pragma unroll
      for (int v = 0; v < 8; ++v) {
        out[(size_t)(gm + v) * kP + gn] = acc[mi][ni][v];
      }
    }
  }
}

// ---------------------------------------------------------------------------
// Plain GEMM:  out[m][p] (+)= sum_c apk[m][c] * bsrc[c][p]
//   apk : [256][Cin] bf16, bsrc : [Cin][P] bf16, out : [256][P] f32
// ---------------------------------------------------------------------------
__global__ __launch_bounds__(128)
void gemm_wmma_kernel(const bf16* __restrict__ apk,
                      const bf16* __restrict__ bsrc,
                      float* __restrict__ out,
                      int Cin, int accFlag)
{
  __shared__ bf16 smem[2 * 64 * kPitch];
  bf16* aTp = smem;
  bf16* bTp = smem + 64 * kPitch;

  const int tid   = threadIdx.x;
  const int wv    = tid >> 5;
  const int lane  = tid & 31;
  const int half  = lane >> 4;
  const int lan15 = lane & 15;

  const int n0 = blockIdx.x * 64;
  const int m0 = blockIdx.y * 64;

  const int wM = (wv >> 1) * 32;
  const int wN = (wv & 1) * 32;

  v8f acc[2][2];
#pragma unroll
  for (int i = 0; i < 2; ++i)
#pragma unroll
    for (int j = 0; j < 2; ++j) acc[i][j] = vzero8();

  const int mRow = tid >> 1;
  const int kSeg = (tid & 1) * 16;
  const int kB_  = tid >> 2;
  const int nA   = (tid & 3) * 16;

  for (int kc = 0; kc < Cin; kc += 32) {
    // ---- stage A via async global->LDS ----
    {
      const bf16* s = apk + ((size_t)(m0 + mRow) * Cin + kc + kSeg);
      async_copy32((unsigned)((mRow * kPitch + kSeg) * 2), s);
    }
    // ---- stage B: vector load + register transpose into [n][k] ----
    {
      const bf16* s = bsrc + ((size_t)(kc + kB_) * kP + n0 + nA);
      bf16 tmp[16];
      *reinterpret_cast<uint4*>(&tmp[0]) = reinterpret_cast<const uint4*>(s)[0];
      *reinterpret_cast<uint4*>(&tmp[8]) = reinterpret_cast<const uint4*>(s)[1];
#pragma unroll
      for (int e = 0; e < 16; ++e) bTp[(nA + e) * kPitch + kB_] = tmp[e];
      if (kc + 32 < Cin)
        __builtin_prefetch((const void*)(s + (size_t)32 * kP), 0, 1);
    }
    async_wait0();
    __syncthreads();

    v16bf af[2], bfg[2];
#pragma unroll
    for (int mi = 0; mi < 2; ++mi) {
      const int mrow = wM + mi * 16 + lan15;
      FragU u;
      u.q[0] = *reinterpret_cast<const uint4*>(&aTp[mrow * kPitch + half * 8]);
      u.q[1] = *reinterpret_cast<const uint4*>(&aTp[mrow * kPitch + 16 + half * 8]);
      af[mi] = u.v;
    }
#pragma unroll
    for (int ni = 0; ni < 2; ++ni) {
      const int ncol = wN + ni * 16 + lan15;
      FragU u;
      u.q[0] = *reinterpret_cast<const uint4*>(&bTp[ncol * kPitch + half * 8]);
      u.q[1] = *reinterpret_cast<const uint4*>(&bTp[ncol * kPitch + 16 + half * 8]);
      bfg[ni] = u.v;
    }
#pragma unroll
    for (int mi = 0; mi < 2; ++mi)
#pragma unroll
      for (int ni = 0; ni < 2; ++ni)
        acc[mi][ni] = __builtin_amdgcn_wmma_f32_16x16x32_bf16(
            false, af[mi], false, bfg[ni], (short)0, acc[mi][ni], false, false);
    __syncthreads();
  }

#pragma unroll
  for (int mi = 0; mi < 2; ++mi) {
#pragma unroll
    for (int ni = 0; ni < 2; ++ni) {
      const int gn = n0 + wN + ni * 16 + lan15;
      const int gm = m0 + wM + mi * 16 + half * 8;
#pragma unroll
      for (int v = 0; v < 8; ++v) {
        const size_t idx = (size_t)(gm + v) * kP + gn;
        float prev = accFlag ? out[idx] : 0.f;
        out[idx] = prev + acc[mi][ni][v];
      }
    }
  }
}

// ---------------------------------------------------------------------------
// Misc kernels
// ---------------------------------------------------------------------------
__global__ void cast_x_kernel(const float* __restrict__ x, bf16* __restrict__ xb)
{
  size_t i = (size_t)blockIdx.x * blockDim.x + threadIdx.x;
  if (i >= kConvOut) return;
  int b  = (int)(i >> 22);          // C*HW = 2^22
  int c  = (int)((i >> 14) & 255);
  int yx = (int)(i & (kHW - 1));
  xb[(size_t)c * kP + ((size_t)b << 14) + yx] = (bf16)x[i];
}

// dst[(t*Mpad + o)*I + ci] = w[((o*I)+ci)*9 + t]  (zero-padded for o >= O)
__global__ void pack3x3_kernel(const float* __restrict__ w, bf16* __restrict__ dst,
                               int O, int I, int Mpad)
{
  size_t n = (size_t)9 * Mpad * I;
  size_t i = (size_t)blockIdx.x * blockDim.x + threadIdx.x;
  if (i >= n) return;
  int ci = (int)(i % I);
  int o  = (int)((i / I) % Mpad);
  int t  = (int)(i / ((size_t)I * Mpad));
  float v = (o < O) ? w[((size_t)o * I + ci) * 9 + t] : 0.f;
  dst[i] = (bf16)v;
}

// 1x1 conv weights are already [O][I]; just cast.
__global__ void pack1x1_kernel(const float* __restrict__ w, bf16* __restrict__ dst)
{
  size_t i = (size_t)blockIdx.x * blockDim.x + threadIdx.x;
  if (i >= (size_t)256 * 256) return;
  dst[i] = (bf16)w[i];
}

__global__ void bn_stats_kernel(const float* __restrict__ t,
                                float* __restrict__ mean, float* __restrict__ rsig)
{
  const int c = blockIdx.x;
  const float* src = t + (size_t)c * kP;
  float s = 0.f, ss = 0.f;
  for (int i = threadIdx.x; i < kP; i += blockDim.x) {
    float v = src[i];
    s += v; ss += v * v;
  }
  __shared__ float sh1[256], sh2[256];
  sh1[threadIdx.x] = s; sh2[threadIdx.x] = ss;
  __syncthreads();
  for (int o = 128; o > 0; o >>= 1) {
    if (threadIdx.x < o) { sh1[threadIdx.x] += sh1[threadIdx.x + o];
                           sh2[threadIdx.x] += sh2[threadIdx.x + o]; }
    __syncthreads();
  }
  if (threadIdx.x == 0) {
    float mu  = sh1[0] / (float)kP;
    float var = sh2[0] / (float)kP - mu * mu;
    mean[c] = mu;
    rsig[c] = rsqrtf(var + kEps);
  }
}

__global__ void bnrelu_bf16_kernel(const float* __restrict__ t,
                                   const float* __restrict__ mu,
                                   const float* __restrict__ rs,
                                   const float* __restrict__ g,
                                   const float* __restrict__ b,
                                   bf16* __restrict__ dst)
{
  size_t i = (size_t)blockIdx.x * blockDim.x + threadIdx.x;
  if (i >= (size_t)kC * kP) return;
  int c = (int)(i >> 15);
  float v = (t[i] - mu[c]) * rs[c] * g[c] + b[c];
  dst[i] = (bf16)fmaxf(v, 0.f);
}

__global__ void cast_f32_bf16_kernel(const float* __restrict__ src, bf16* __restrict__ dst)
{
  size_t i = (size_t)blockIdx.x * blockDim.x + threadIdx.x;
  if (i >= (size_t)kC * kP) return;
  dst[i] = (bf16)src[i];
}

__global__ void init_bias_kernel(float* __restrict__ psum,
                                 const float* __restrict__ b1,
                                 const float* __restrict__ b2)
{
  size_t i = (size_t)blockIdx.x * blockDim.x + threadIdx.x;
  if (i >= (size_t)kC * kP) return;
  int c = (int)(i >> 15);
  psum[i] = b1[c] + b2[c];
}

// Offset / mask transform. offraw: [64][P] f32 (27 valid rows, pre-bias).
__global__ void off_transform_kernel(const float* __restrict__ offraw,
                                     const float* __restrict__ offb,
                                     float* __restrict__ doff,   // d_out + kConvOut
                                     float* __restrict__ maskb,
                                     int boff, int mode)
{
  int p = blockIdx.x * blockDim.x + threadIdx.x;
  if (p >= kP) return;
  int b  = p >> 14;
  int yx = p & (kHW - 1);
#pragma unroll
  for (int k = 0; k < 9; ++k) {
    float dx = offraw[(size_t)(2 * k + 1) * kP + p] + offb[2 * k + 1];
    float mr = offraw[(size_t)(18 + k) * kP + p]    + offb[18 + k];
    float sig = 1.f / (1.f + expf(-mr));
    bool keep = (mode == 0) ? (dx >= 0.f) : (dx <= 0.f);
    maskb[(size_t)k * kP + p] = keep ? sig : 0.f;
    doff[((size_t)b * 36 + boff + 2 * k)     * kHW + yx] = 0.0f;  // dy zeroed
    doff[((size_t)b * 36 + boff + 2 * k + 1) * kHW + yx] = dx;
  }
}

// Bilinear col build for one tap: col[c][p] = bilinear(pb, py, px) * mask[k][p]
__global__ void col_build_kernel(const bf16* __restrict__ pb,
                                 const float* __restrict__ doff,
                                 const float* __restrict__ maskb,
                                 bf16* __restrict__ col,
                                 int boff, int k)
{
  int p = blockIdx.x * blockDim.x + threadIdx.x;
  if (p >= kP) return;
  int b  = p >> 14;
  int yx = p & (kHW - 1);
  int y  = yx >> 7;
  int x  = yx & (kW - 1);
  int ki = k / 3, kj = k % 3;

  float dy = doff[((size_t)b * 36 + boff + 2 * k)     * kHW + yx];
  float dx = doff[((size_t)b * 36 + boff + 2 * k + 1) * kHW + yx];
  float pyf = (float)(y + ki - 1) + dy;
  float pxf = (float)(x + kj - 1) + dx;
  float m = maskb[(size_t)k * kP + p];

  float y0f = floorf(pyf), x0f = floorf(pxf);
  float ly = pyf - y0f, lx = pxf - x0f;
  int y0 = (int)y0f, x0i = (int)x0f;
  int y1 = y0 + 1,   x1 = x0i + 1;

  float w00 = (1.f - ly) * (1.f - lx);
  float w01 = (1.f - ly) * lx;
  float w10 = ly * (1.f - lx);
  float w11 = ly * lx;

  bool v00 = (y0 >= 0 && y0 < kH && x0i >= 0 && x0i < kW);
  bool v01 = (y0 >= 0 && y0 < kH && x1  >= 0 && x1  < kW);
  bool v10 = (y1 >= 0 && y1 < kH && x0i >= 0 && x0i < kW);
  bool v11 = (y1 >= 0 && y1 < kH && x1  >= 0 && x1  < kW);

  int yc0 = y0  < 0 ? 0 : (y0  > kH - 1 ? kH - 1 : y0);
  int yc1 = y1  < 0 ? 0 : (y1  > kH - 1 ? kH - 1 : y1);
  int xc0 = x0i < 0 ? 0 : (x0i > kW - 1 ? kW - 1 : x0i);
  int xc1 = x1  < 0 ? 0 : (x1  > kW - 1 ? kW - 1 : x1);

  int base = b << 14;
  int i00 = base + (yc0 << 7) + xc0;
  int i01 = base + (yc0 << 7) + xc1;
  int i10 = base + (yc1 << 7) + xc0;
  int i11 = base + (yc1 << 7) + xc1;

  for (int c = 0; c < kC; ++c) {
    const bf16* row = pb + (size_t)c * kP;
    float a = 0.f;
    if (v00) a += w00 * (float)row[i00];
    if (v01) a += w01 * (float)row[i01];
    if (v10) a += w10 * (float)row[i10];
    if (v11) a += w11 * (float)row[i11];
    col[(size_t)c * kP + p] = (bf16)(a * m);
  }
}

__global__ void fuse_relu1_kernel(const float* __restrict__ t3, const float* __restrict__ t4,
                                  const float* __restrict__ mu3, const float* __restrict__ rs3,
                                  const float* __restrict__ g3,  const float* __restrict__ b3,
                                  const float* __restrict__ mu4, const float* __restrict__ rs4,
                                  const float* __restrict__ g4,  const float* __restrict__ b4,
                                  bf16* __restrict__ dst)
{
  size_t i = (size_t)blockIdx.x * blockDim.x + threadIdx.x;
  if (i >= (size_t)kC * kP) return;
  int c = (int)(i >> 15);
  float a = (t3[i] - mu3[c]) * rs3[c] * g3[c] + b3[c];
  float b = (t4[i] - mu4[c]) * rs4[c] * g4[c] + b4[c];
  dst[i] = (bf16)fmaxf(a + b, 0.f);
}

__global__ void final_out_kernel(const float* __restrict__ t5,
                                 const float* __restrict__ mu, const float* __restrict__ rs,
                                 const float* __restrict__ g,  const float* __restrict__ b,
                                 float* __restrict__ out)
{
  size_t i = (size_t)blockIdx.x * blockDim.x + threadIdx.x;
  if (i >= (size_t)kC * kP) return;
  int c  = (int)(i >> 15);
  int p  = (int)(i & (kP - 1));
  int bb = p >> 14;
  int yx = p & (kHW - 1);
  float v = fmaxf((t5[i] - mu[c]) * rs[c] * g[c] + b[c], 0.f);
  out[((size_t)bb * kC + c) * kHW + yx] = v;
}

}  // namespace dcnpool

// ---------------------------------------------------------------------------
extern "C" void kernel_launch(void* const* d_in, const int* in_sizes, int n_in,
                              void* d_out, int out_size, void* d_ws, size_t ws_size,
                              hipStream_t stream)
{
  using namespace dcnpool;
  (void)in_sizes; (void)n_in; (void)out_size;

  const float* x   = (const float*)d_in[0];
  const float* p1w = (const float*)d_in[1];
  const float* p1g = (const float*)d_in[2];
  const float* p1b = (const float*)d_in[3];
  const float* p2w = (const float*)d_in[4];
  const float* p2g = (const float*)d_in[5];
  const float* p2b = (const float*)d_in[6];
  const float* o1w = (const float*)d_in[7];
  const float* o1b = (const float*)d_in[8];
  const float* q1w = (const float*)d_in[9];
  const float* q1b = (const float*)d_in[10];
  const float* o2w = (const float*)d_in[11];
  const float* o2b = (const float*)d_in[12];
  const float* q2w = (const float*)d_in[13];
  const float* q2b = (const float*)d_in[14];
  const float* pcw = (const float*)d_in[15];
  const float* pbg = (const float*)d_in[16];
  const float* pbb = (const float*)d_in[17];
  const float* c1w = (const float*)d_in[18];
  const float* b1g = (const float*)d_in[19];
  const float* b1b = (const float*)d_in[20];
  const float* c2w = (const float*)d_in[21];
  const float* c2g = (const float*)d_in[22];
  const float* c2b = (const float*)d_in[23];

  char* ws = (char*)d_ws;
  size_t off = 0;
  auto take = [&](size_t bytes) -> char* {
    char* p = ws + off;
    off = (off + bytes + 255) & ~(size_t)255;
    return p;
  };

  bf16*  xb    = (bf16*) take((size_t)kC * kP * 2);
  bf16*  pb1   = (bf16*) take((size_t)kC * kP * 2);
  bf16*  pb2   = (bf16*) take((size_t)kC * kP * 2);
  float* tbuf  = (float*)take((size_t)kC * kP * 4);
  float* psum  = (float*)take((size_t)kC * kP * 4);
  bf16*  colb  = (bf16*) take((size_t)kC * kP * 2);
  float* offr  = (float*)take((size_t)64 * kP * 4);
  float* maskb = (float*)take((size_t)9 * kP * 4);
  bf16*  wp1   = (bf16*) take((size_t)9 * 256 * 256 * 2);
  bf16*  wp2   = (bf16*) take((size_t)9 * 256 * 256 * 2);
  bf16*  wq1   = (bf16*) take((size_t)9 * 256 * 256 * 2);
  bf16*  wq2   = (bf16*) take((size_t)9 * 256 * 256 * 2);
  bf16*  wpc   = (bf16*) take((size_t)9 * 256 * 256 * 2);
  bf16*  wc2   = (bf16*) take((size_t)9 * 256 * 256 * 2);
  bf16*  wc1   = (bf16*) take((size_t)256 * 256 * 2);
  bf16*  wo1   = (bf16*) take((size_t)9 * 64 * 256 * 2);
  bf16*  wo2   = (bf16*) take((size_t)9 * 64 * 256 * 2);
  float* muA   = (float*)take(256 * 4);
  float* rsA   = (float*)take(256 * 4);
  float* muB   = (float*)take(256 * 4);
  float* rsB   = (float*)take(256 * 4);
  if (off > ws_size) return;  // insufficient scratch: deterministic no-op

  float* doutF = (float*)d_out;
  float* doff  = doutF + kConvOut;

  const dim3 blk256(256);
  const dim3 gEW((unsigned)((kConvOut + 255) / 256));       // C*P elementwise
  const dim3 gP((unsigned)((kP + 255) / 256));              // per-pixel
  const dim3 blk128(128);
  const dim3 gConv(kP / 64, kC / 64);                       // (512, 4)
  const dim3 gConvOff(kP / 64, 1);                          // (512, 1)

  // --- packing / casting ---
  {
    size_t n = (size_t)9 * 256 * 256;
    dim3 g((unsigned)((n + 255) / 256));
    pack3x3_kernel<<<g, blk256, 0, stream>>>(p1w, wp1, 256, 256, 256);
    pack3x3_kernel<<<g, blk256, 0, stream>>>(p2w, wp2, 256, 256, 256);
    pack3x3_kernel<<<g, blk256, 0, stream>>>(q1w, wq1, 256, 256, 256);
    pack3x3_kernel<<<g, blk256, 0, stream>>>(q2w, wq2, 256, 256, 256);
    pack3x3_kernel<<<g, blk256, 0, stream>>>(pcw, wpc, 256, 256, 256);
    pack3x3_kernel<<<g, blk256, 0, stream>>>(c2w, wc2, 256, 256, 256);
  }
  {
    size_t n = (size_t)9 * 64 * 256;
    dim3 g((unsigned)((n + 255) / 256));
    pack3x3_kernel<<<g, blk256, 0, stream>>>(o1w, wo1, 27, 256, 64);
    pack3x3_kernel<<<g, blk256, 0, stream>>>(o2w, wo2, 27, 256, 64);
  }
  pack1x1_kernel<<<dim3((256 * 256 + 255) / 256), blk256, 0, stream>>>(c1w, wc1);
  cast_x_kernel<<<gEW, blk256, 0, stream>>>(x, xb);

  // --- p1 = relu(bn(conv(x, p1_w))) ---
  conv3x3_wmma_kernel<<<gConv, blk128, 0, stream>>>(wp1, xb, tbuf, 256, 256);
  bn_stats_kernel<<<dim3(kC), blk256, 0, stream>>>(tbuf, muA, rsA);
  bnrelu_bf16_kernel<<<gEW, blk256, 0, stream>>>(tbuf, muA, rsA, p1g, p1b, pb1);

  // --- p2 = relu(bn(conv(x, p2_w))) ---
  conv3x3_wmma_kernel<<<gConv, blk128, 0, stream>>>(wp2, xb, tbuf, 256, 256);
  bn_stats_kernel<<<dim3(kC), blk256, 0, stream>>>(tbuf, muA, rsA);
  bnrelu_bf16_kernel<<<gEW, blk256, 0, stream>>>(tbuf, muA, rsA, p2g, p2b, pb2);

  // --- psum = bias1 + bias2 (DCN accumulators for pool1 + pool2) ---
  init_bias_kernel<<<gEW, blk256, 0, stream>>>(psum, q1b, q2b);

  // --- DCN branch 1 (mode 0) ---
  conv3x3_wmma_kernel<<<gConvOff, blk128, 0, stream>>>(wo1, pb1, offr, 256, 64);
  off_transform_kernel<<<gP, blk256, 0, stream>>>(offr, o1b, doff, maskb, 0, 0);
  for (int k = 0; k < 9; ++k) {
    col_build_kernel<<<gP, blk256, 0, stream>>>(pb1, doff, maskb, colb, 0, k);
    gemm_wmma_kernel<<<gConv, blk128, 0, stream>>>(wq1 + (size_t)k * 256 * 256,
                                                   colb, psum, 256, 1);
  }

  // --- DCN branch 2 (mode 2) ---
  conv3x3_wmma_kernel<<<gConvOff, blk128, 0, stream>>>(wo2, pb2, offr, 256, 64);
  off_transform_kernel<<<gP, blk256, 0, stream>>>(offr, o2b, doff, maskb, 18, 2);
  for (int k = 0; k < 9; ++k) {
    col_build_kernel<<<gP, blk256, 0, stream>>>(pb2, doff, maskb, colb, 18, k);
    gemm_wmma_kernel<<<gConv, blk128, 0, stream>>>(wq2 + (size_t)k * 256 * 256,
                                                   colb, psum, 256, 1);
  }

  // --- p_bn1 = bn(conv(pool1+pool2, pc_w)) ; t3 in tbuf, stats (muA,rsA) ---
  cast_f32_bf16_kernel<<<gEW, blk256, 0, stream>>>(psum, colb);
  conv3x3_wmma_kernel<<<gConv, blk128, 0, stream>>>(wpc, colb, tbuf, 256, 256);
  bn_stats_kernel<<<dim3(kC), blk256, 0, stream>>>(tbuf, muA, rsA);

  // --- bn1 = bn(conv1x1(x, c1_w)) ; t4 in psum, stats (muB,rsB) ---
  gemm_wmma_kernel<<<gConv, blk128, 0, stream>>>(wc1, xb, psum, 256, 0);
  bn_stats_kernel<<<dim3(kC), blk256, 0, stream>>>(psum, muB, rsB);

  // --- relu1 = relu(p_bn1 + bn1) -> pb1 (bf16) ---
  fuse_relu1_kernel<<<gEW, blk256, 0, stream>>>(tbuf, psum,
                                                muA, rsA, pbg, pbb,
                                                muB, rsB, b1g, b1b, pb1);

  // --- conv2 = relu(bn(conv(relu1, c2_w))) -> d_out conv region ---
  conv3x3_wmma_kernel<<<gConv, blk128, 0, stream>>>(wc2, pb1, tbuf, 256, 256);
  bn_stats_kernel<<<dim3(kC), blk256, 0, stream>>>(tbuf, muA, rsA);
  final_out_kernel<<<gEW, blk256, 0, stream>>>(tbuf, muA, rsA, c2g, c2b, doutF);
}
